// RWKV_TimeMix_x051a_32856499814863
// MI455X (gfx1250) — matrix operand order repaired and checked by reading, not compiled, not run on hardware
//
#include <hip/hip_runtime.h>

// ---------------------------------------------------------------------------
// Types for CDNA5 WMMA
// ---------------------------------------------------------------------------
typedef __attribute__((ext_vector_type(16))) __bf16 v16bf;
typedef __attribute__((ext_vector_type(8)))  __bf16 v8bf;
typedef __attribute__((ext_vector_type(8)))  float  v8f;
typedef __attribute__((ext_vector_type(2)))  float  v2f;
typedef __attribute__((ext_vector_type(4)))  int    v4i;

#define AS1 __attribute__((address_space(1)))
#define AS3 __attribute__((address_space(3)))

#define WMMA_BF16(a, b, c) \
  __builtin_amdgcn_wmma_f32_16x16x32_bf16(false, (a), false, (b), (short)0, (c), false, false)

#if __has_builtin(__builtin_amdgcn_wmma_f32_16x16x4_f32)
#define WMMA_F32x4(a, b, c) \
  __builtin_amdgcn_wmma_f32_16x16x4_f32(false, (a), false, (b), (short)0, (c), false, false)
#else
#warning "builtin __builtin_amdgcn_wmma_f32_16x16x4_f32 missing: emulation fallback active"
static __device__ __forceinline__ v8f wmma4_emul(v2f a, v2f b, v8f c) {
#pragma unroll
  for (int i = 0; i < 8; ++i) c[i] += a[0] * b[0] + a[1] * b[1];
  return c;
}
#define WMMA_F32x4(a, b, c) wmma4_emul((a), (b), (c))
#endif

#if __has_builtin(__builtin_amdgcn_global_load_async_to_lds_b128) && \
    __has_builtin(__builtin_amdgcn_s_wait_asynccnt)
#define HAVE_ASYNC_LDS 1
#else
#define HAVE_ASYNC_LDS 0
#warning "async-to-LDS builtins missing: synchronous LDS staging fallback active"
#endif

static __device__ __forceinline__ v8f zero8() {
  v8f v;
#pragma unroll
  for (int i = 0; i < 8; ++i) v[i] = 0.f;
  return v;
}
static __device__ __forceinline__ v2f mk2(float a, float b) {
  v2f t; t[0] = a; t[1] = b; return t;
}
// A fragment (16x32 bf16): lanes<16 hold K 0..7 & 16..23; lanes>=16 K 8..15 & 24..31.
static __device__ __forceinline__ v16bf frag_a(const __bf16* base) {
  v8bf lo = *(const v8bf*)(base);
  v8bf hi = *(const v8bf*)(base + 16);
  v16bf r;
#pragma unroll
  for (int e = 0; e < 8; ++e) { r[e] = lo[e]; r[e + 8] = hi[e]; }
  return r;
}
// B fragment (32x16 bf16 from W rows): lanes<16 hold K 0..7 & 16..23; lanes>=16 K 8..15 & 24..31.
static __device__ __forceinline__ v16bf frag_b(const __bf16* base) {
  v8bf lo = *(const v8bf*)(base);
  v8bf hi = *(const v8bf*)(base + 16);
  v16bf r;
#pragma unroll
  for (int e = 0; e < 8; ++e) { r[e] = lo[e]; r[e + 8] = hi[e]; }
  return r;
}

// ---------------------------------------------------------------------------
// Kernel 1: fp32 -> bf16 convert (weights)
// ---------------------------------------------------------------------------
__global__ void cvt_bf16_kernel(const float* __restrict__ in, __bf16* __restrict__ out,
                                long long n) {
  long long i = (long long)blockIdx.x * blockDim.x + threadIdx.x;
  if (i < n) out[i] = (__bf16)in[i];
}

// ---------------------------------------------------------------------------
// Kernel 2: token shift mix -> bf16 activations xk/xv/xr/xg
// ---------------------------------------------------------------------------
__global__ void mix_kernel(const float* __restrict__ x,
                           const float* __restrict__ mk, const float* __restrict__ mv,
                           const float* __restrict__ mr, const float* __restrict__ mg,
                           __bf16* __restrict__ xk, __bf16* __restrict__ xv,
                           __bf16* __restrict__ xr, __bf16* __restrict__ xg,
                           int Tn, int Cn, long long n) {
  long long i = (long long)blockIdx.x * blockDim.x + threadIdx.x;
  if (i >= n) return;
  int cc = (int)(i % Cn);
  long long bt = i / Cn;
  int t = (int)(bt % Tn);
  float xc = x[i];
  float xp = (t > 0) ? x[i - Cn] : 0.f;
  float xx = xp - xc;
  xk[i] = (__bf16)(xc + xx * mk[cc]);
  xv[i] = (__bf16)(xc + xx * mv[cc]);
  xr[i] = (__bf16)(xc + xx * mr[cc]);
  xg[i] = (__bf16)(xc + xx * mg[cc]);
}

// ---------------------------------------------------------------------------
// Kernel 3: bf16 GEMM  out[M,N] = A[M,K] @ W[N,K]^T   (f32 accumulate)
// Double-buffered LDS; tiles prefetched with GLOBAL_LOAD_ASYNC_TO_LDS_B128
// (ASYNCcnt-tracked) when available.
// mode 0: f32 output permuted to [B,H,T,64]  (for r/k/v)
// mode 1: silu -> bf16, row-major [M,N]      (for g)
// mode 2: f32 row-major [M,N]                (final output)
// ---------------------------------------------------------------------------
__global__ __launch_bounds__(256) void gemm_bf16_kernel(
    const __bf16* __restrict__ A, const __bf16* __restrict__ W,
    float* __restrict__ outF, __bf16* __restrict__ outB,
    int M, int N, int K, int mode, int Hh, int Tt) {
  constexpr int AP = 40;  // LDS row pitch in bf16 elems (80B -> conflict-free b128 reads)
  __shared__ __bf16 lA[2][128 * AP];
  __shared__ __bf16 lB[2][128 * AP];
  const int tid  = threadIdx.x;
  const int wave = tid >> 5;
  const int lane = tid & 31;
  const int hl   = lane >> 4;   // lane half
  const int r16  = lane & 15;
  const int bm = blockIdx.x * 128;
  const int bn = blockIdx.y * 128;
  const int m0 = (wave >> 2) * 64;  // wave computes 64x32
  const int n0 = (wave & 3) * 32;

  // each thread moves 4 x 16B per tile (2 for A, 2 for B)
  const int ldrow0 = tid >> 2;
  const int ldcol  = (tid & 3) * 8;

  auto issue_tile = [&](int buf, int kb) {
#pragma unroll
    for (int it = 0; it < 2; ++it) {
      int row = ldrow0 + it * 64;
      const __bf16* gA = &A[(size_t)(bm + row) * K + kb + ldcol];
      const __bf16* gW = &W[(size_t)(bn + row) * K + kb + ldcol];
      __bf16* sA = &lA[buf][row * AP + ldcol];
      __bf16* sB = &lB[buf][row * AP + ldcol];
#if HAVE_ASYNC_LDS
      __builtin_amdgcn_global_load_async_to_lds_b128(
          (AS1 v4i*)gA, (AS3 v4i*)sA, 0, 0);
      __builtin_amdgcn_global_load_async_to_lds_b128(
          (AS1 v4i*)gW, (AS3 v4i*)sB, 0, 0);
#else
      *(v8bf*)sA = *(const v8bf*)gA;
      *(v8bf*)sB = *(const v8bf*)gW;
#endif
    }
  };

  v8f acc[4][2];
#pragma unroll
  for (int i = 0; i < 4; ++i)
#pragma unroll
    for (int j = 0; j < 2; ++j) acc[i][j] = zero8();

  const int NKB = K >> 5;  // K/32 tiles
  issue_tile(0, 0);
  if (NKB > 1) issue_tile(1, 32);

  for (int ib = 0; ib < NKB; ++ib) {
    const int cur = ib & 1;
#if HAVE_ASYNC_LDS
    if (ib + 1 < NKB) __builtin_amdgcn_s_wait_asynccnt(4);  // tile ib retired
    else              __builtin_amdgcn_s_wait_asynccnt(0);
#endif
    __syncthreads();
    v16bf aF[4], bF[2];
#pragma unroll
    for (int i = 0; i < 4; ++i)
      aF[i] = frag_a(&lA[cur][(m0 + i * 16 + r16) * AP + hl * 8]);
#pragma unroll
    for (int j = 0; j < 2; ++j)
      bF[j] = frag_b(&lB[cur][(n0 + j * 16 + r16) * AP + hl * 8]);
#pragma unroll
    for (int i = 0; i < 4; ++i)
#pragma unroll
      for (int j = 0; j < 2; ++j) acc[i][j] = WMMA_BF16(aF[i], bF[j], acc[i][j]);
    __syncthreads();  // all waves done reading buf[cur] before refill
    if (ib + 2 < NKB) issue_tile(cur, (ib + 2) * 32);
  }

  // epilogue: C layout = lane (N), vgpr e -> M = e + 8*half
#pragma unroll
  for (int i = 0; i < 4; ++i)
#pragma unroll
    for (int j = 0; j < 2; ++j)
#pragma unroll
      for (int e = 0; e < 8; ++e) {
        int m = bm + m0 + i * 16 + hl * 8 + e;
        int n = bn + n0 + j * 16 + r16;
        float v = acc[i][j][e];
        if (mode == 0) {
          int bb = m / Tt, t = m - bb * Tt;
          int hh = n >> 6, nn = n & 63;
          outF[(((size_t)bb * Hh + hh) * Tt + t) * 64 + nn] = v;
        } else if (mode == 1) {
          float s = v / (1.f + __expf(-v));  // silu
          outB[(size_t)m * N + n] = (__bf16)s;
        } else {
          outF[(size_t)m * N + n] = v;
        }
      }
}

// ---------------------------------------------------------------------------
// Kernel 4: chunked linear attention (Q=256), f32 WMMA 16x16x4
// r/k/v/y all [B,H,T,64] f32. One block (8 waves) per (b,h); state in LDS.
// Decay powers w^t precomputed in a 257-entry LDS table (replaces per-element
// transcendentals in the WMMA hot loops).
// ---------------------------------------------------------------------------
__global__ __launch_bounds__(256) void rwkv_attn_kernel(
    const float* __restrict__ rg, const float* __restrict__ kg,
    const float* __restrict__ vg, const float* __restrict__ td,
    const float* __restrict__ tf, float* __restrict__ yg,
    int Bn, int Hn, int Tn) {
  constexpr int NQ = 256;
  __shared__ float st[64][66];          // state[n_k][n_v], padded
  __shared__ float stage[8][16][18];    // per-wave att-tile transpose staging
  __shared__ float wpow[260];           // w^t, t = 0..256
  const int bh   = blockIdx.x;
  const int b    = bh / Hn;
  const int h    = bh - b * Hn;
  const int tid  = threadIdx.x;
  const int wave = tid >> 5;
  const int lane = tid & 31;
  const int hl   = lane >> 4;
  const int ln   = lane & 15;

  const float logw = -__expf(td[h]);    // log(w) = -exp(time_decay)
  const float u    = tf[h];
  const size_t base = ((size_t)b * Hn + h) * (size_t)Tn * 64;

  for (int i = tid; i < 64 * 66; i += 256) ((float*)st)[i] = 0.f;
  for (int i = tid; i < 257; i += 256) wpow[i] = __expf(logw * (float)i);
  __syncthreads();

  const int nChunks = Tn / NQ;
  for (int c = 0; c < nChunks; ++c) {
    const int t0 = c * NQ;
    // -------- phase 1: y for this chunk (reads state) --------
    for (int rt2 = 0; rt2 < 2; ++rt2) {
      const int i0 = (wave * 2 + rt2) * 16;
      // preload r rows: A layout, K=64 in 16 steps of 4
      v2f rA[16];
      const float* rrow = rg + base + (size_t)(t0 + i0 + ln) * 64;
#pragma unroll
      for (int k4 = 0; k4 < 16; ++k4) {
        int k = k4 * 4 + 2 * hl;
        rA[k4] = mk2(rrow[k], rrow[k + 1]);
      }
      v8f acc[4];
      // term: (rc @ state) * wb  ; wb = w^i
#pragma unroll
      for (int nt = 0; nt < 4; ++nt) {
        v8f yv = zero8();
#pragma unroll
        for (int k4 = 0; k4 < 16; ++k4) {
          int k = k4 * 4 + 2 * hl;
          v2f bB = mk2(st[k][nt * 16 + ln], st[k + 1][nt * 16 + ln]);
          yv = WMMA_F32x4(rA[k4], bB, yv);
        }
#pragma unroll
        for (int e = 0; e < 8; ++e) yv[e] *= wpow[i0 + e + 8 * hl];
        acc[nt] = yv;
      }
      // term: (rc @ kc^T * w_mat) @ vc, streamed over j-tiles
      for (int jt = 0; jt < 16; ++jt) {
        v8f att = zero8();
        const float* krow = kg + base + (size_t)(t0 + jt * 16 + ln) * 64;
#pragma unroll
        for (int k4 = 0; k4 < 16; ++k4) {
          int k = k4 * 4 + 2 * hl;
          v2f bB = mk2(krow[k], krow[k + 1]);
          att = WMMA_F32x4(rA[k4], bB, att);
        }
        // w_mat mask: i>j -> w^(i-j-1); i==j -> u; i<j -> 0
#pragma unroll
        for (int e = 0; e < 8; ++e) {
          int i = i0 + e + 8 * hl;
          int j = jt * 16 + ln;
          int d = i - j;
          float f = (d > 0) ? wpow[d - 1] : ((d == 0) ? u : 0.f);
          att[e] *= f;
        }
        // C-layout -> row-major staging (same-wave LDS, in order)
#pragma unroll
        for (int e = 0; e < 8; ++e) stage[wave][e + 8 * hl][ln] = att[e];
        // acc += att16 @ vc[jt-rows]
#pragma unroll
        for (int k4 = 0; k4 < 4; ++k4) {
          int kk = k4 * 4 + 2 * hl;
          v2f aA = mk2(stage[wave][ln][kk], stage[wave][ln][kk + 1]);
          const float* vrow0 = vg + base + (size_t)(t0 + jt * 16 + kk) * 64;
          const float* vrow1 = vrow0 + 64;
#pragma unroll
          for (int nt = 0; nt < 4; ++nt) {
            v2f bB = mk2(vrow0[nt * 16 + ln], vrow1[nt * 16 + ln]);
            acc[nt] = WMMA_F32x4(aA, bB, acc[nt]);
          }
        }
      }
      // store y tile strip
#pragma unroll
      for (int nt = 0; nt < 4; ++nt)
#pragma unroll
        for (int e = 0; e < 8; ++e) {
          int i = i0 + e + 8 * hl;
          yg[base + (size_t)(t0 + i) * 64 + nt * 16 + ln] = acc[nt][e];
        }
    }
    __syncthreads();  // all state reads done
    // -------- phase 2: state = w^Q * state + (kc * wk) @ vc ; wk = w^(Q-1-t)
    const float wsf = wpow[256];
    for (int tt2 = 0; tt2 < 2; ++tt2) {
      const int tile = wave * 2 + tt2;  // 16 tiles of 16x16
      const int at = tile >> 2;         // n_k tile
      const int bt = tile & 3;          // n_v tile
      v8f s = zero8();
#pragma unroll 4
      for (int k4 = 0; k4 < 64; ++k4) {
        int t = k4 * 4 + 2 * hl;
        v2f aA = mk2(kg[base + (size_t)(t0 + t) * 64 + at * 16 + ln] * wpow[NQ - 1 - t],
                     kg[base + (size_t)(t0 + t + 1) * 64 + at * 16 + ln] * wpow[NQ - 2 - t]);
        v2f bB = mk2(vg[base + (size_t)(t0 + t) * 64 + bt * 16 + ln],
                     vg[base + (size_t)(t0 + t + 1) * 64 + bt * 16 + ln]);
        s = WMMA_F32x4(aA, bB, s);
      }
#pragma unroll
      for (int e = 0; e < 8; ++e) {
        int m = at * 16 + e + 8 * hl;
        int n = bt * 16 + ln;
        st[m][n] = wsf * st[m][n] + s[e];
      }
    }
    __syncthreads();
  }
}

// ---------------------------------------------------------------------------
// Kernel 5: per-head groupnorm + gamma/beta + gate by g -> bf16 activation
// ---------------------------------------------------------------------------
__global__ __launch_bounds__(256) void gn_gate_kernel(
    const float* __restrict__ yg, const __bf16* __restrict__ gB,
    const float* __restrict__ gamma, const float* __restrict__ beta,
    __bf16* __restrict__ actO, int Bn, int Hn, int Tn) {
  const int gid  = blockIdx.x * 8 + (threadIdx.x >> 5);  // one wave per (b,t,h)
  const int lane = threadIdx.x & 31;
  const int total = Bn * Tn * Hn;
  if (gid >= total) return;
  const int h = gid % Hn;
  const int t = (gid / Hn) % Tn;
  const int b = gid / (Hn * Tn);
  const float* row = yg + (((size_t)b * Hn + h) * Tn + t) * 64;
  float v0 = row[lane], v1 = row[lane + 32];
  float s = v0 + v1, s2 = v0 * v0 + v1 * v1;
#pragma unroll
  for (int off = 16; off > 0; off >>= 1) {
    s  += __shfl_xor(s, off, 32);
    s2 += __shfl_xor(s2, off, 32);
  }
  const float mu  = s * (1.f / 64.f);
  const float var = s2 * (1.f / 64.f) - mu * mu;
  const float inv = rsqrtf(var + 1.44e-3f);  // EPS = 1e-5 * 144
  const size_t crow = ((size_t)b * Tn + t) * (size_t)(Hn * 64);
  const int c0 = h * 64 + lane;
  float g0 = (float)gB[crow + c0], g1 = (float)gB[crow + c0 + 32];
  actO[crow + c0]      = (__bf16)(((v0 - mu) * inv * gamma[c0]      + beta[c0])      * g0);
  actO[crow + c0 + 32] = (__bf16)(((v1 - mu) * inv * gamma[c0 + 32] + beta[c0 + 32]) * g1);
}

// ---------------------------------------------------------------------------
// Launch
// ---------------------------------------------------------------------------
extern "C" void kernel_launch(void* const* d_in, const int* in_sizes, int n_in,
                              void* d_out, int out_size, void* d_ws, size_t ws_size,
                              hipStream_t stream) {
  (void)in_sizes; (void)n_in; (void)out_size; (void)ws_size;
  const int Bn = 4, Tn = 2048, Cn = 2048, Hn = 32;
  const float* x    = (const float*)d_in[0];
  const float* mk   = (const float*)d_in[1];
  const float* mv   = (const float*)d_in[2];
  const float* mr   = (const float*)d_in[3];
  const float* mg   = (const float*)d_in[4];
  const float* td   = (const float*)d_in[5];
  const float* tf   = (const float*)d_in[6];
  const float* Wr   = (const float*)d_in[7];
  const float* Wk   = (const float*)d_in[8];
  const float* Wv   = (const float*)d_in[9];
  const float* Wg   = (const float*)d_in[10];
  const float* Wo   = (const float*)d_in[11];
  const float* gam  = (const float*)d_in[12];
  const float* bet  = (const float*)d_in[13];

  const size_t M      = (size_t)Bn * Tn;                  // 8192
  const size_t SZ_ACT = M * Cn * sizeof(__bf16);          // 32 MB
  const size_t SZ_W   = (size_t)Cn * Cn * sizeof(__bf16); // 8 MB
  const size_t SZ_F   = M * Cn * sizeof(float);           // 64 MB
  char* p = (char*)d_ws;
  __bf16* xrB = (__bf16*)p; p += SZ_ACT;
  __bf16* xkB = (__bf16*)p; p += SZ_ACT;
  __bf16* xvB = (__bf16*)p; p += SZ_ACT;
  __bf16* xgB = (__bf16*)p; p += SZ_ACT;
  __bf16* wrB = (__bf16*)p; p += SZ_W;
  __bf16* wkB = (__bf16*)p; p += SZ_W;
  __bf16* wvB = (__bf16*)p; p += SZ_W;
  __bf16* wgB = (__bf16*)p; p += SZ_W;
  __bf16* woB = (__bf16*)p; p += SZ_W;
  float*  rg  = (float*)p;  p += SZ_F;
  float*  kg  = (float*)p;  p += SZ_F;
  float*  vg  = (float*)p;  p += SZ_F;
  __bf16* gB  = (__bf16*)p; p += SZ_ACT;
  // aliases (lifetimes disjoint): y over xr+xk region, gated act over xv region
  float*  yg  = (float*)xrB;
  __bf16* aO  = xvB;

  const long long nW = (long long)Cn * Cn;
  cvt_bf16_kernel<<<(nW + 255) / 256, 256, 0, stream>>>(Wr, wrB, nW);
  cvt_bf16_kernel<<<(nW + 255) / 256, 256, 0, stream>>>(Wk, wkB, nW);
  cvt_bf16_kernel<<<(nW + 255) / 256, 256, 0, stream>>>(Wv, wvB, nW);
  cvt_bf16_kernel<<<(nW + 255) / 256, 256, 0, stream>>>(Wg, wgB, nW);
  cvt_bf16_kernel<<<(nW + 255) / 256, 256, 0, stream>>>(Wo, woB, nW);

  const long long nA = (long long)M * Cn;
  mix_kernel<<<(nA + 255) / 256, 256, 0, stream>>>(x, mk, mv, mr, mg,
                                                   xkB, xvB, xrB, xgB, Tn, Cn, nA);

  dim3 gg((unsigned)(M / 128), (unsigned)(Cn / 128));
  gemm_bf16_kernel<<<gg, 256, 0, stream>>>(xrB, wrB, rg, nullptr, (int)M, Cn, Cn, 0, Hn, Tn);
  gemm_bf16_kernel<<<gg, 256, 0, stream>>>(xkB, wkB, kg, nullptr, (int)M, Cn, Cn, 0, Hn, Tn);
  gemm_bf16_kernel<<<gg, 256, 0, stream>>>(xvB, wvB, vg, nullptr, (int)M, Cn, Cn, 0, Hn, Tn);
  gemm_bf16_kernel<<<gg, 256, 0, stream>>>(xgB, wgB, nullptr, gB, (int)M, Cn, Cn, 1, Hn, Tn);

  rwkv_attn_kernel<<<Bn * Hn, 256, 0, stream>>>(rg, kg, vg, td, tf, yg, Bn, Hn, Tn);

  gn_gate_kernel<<<(Bn * Tn * Hn) / 8, 256, 0, stream>>>(yg, gB, gam, bet, aO, Bn, Hn, Tn);

  gemm_bf16_kernel<<<gg, 256, 0, stream>>>(aO, woB, (float*)d_out, nullptr,
                                           (int)M, Cn, Cn, 2, Hn, Tn);
}
